// ResGateConv_normalization_44178033607164
// MI455X (gfx1250) — matrix operand order repaired and checked by reading, not compiled, not run on hardware
//
#include <hip/hip_runtime.h>
#include <hip/hip_bf16.h>

#define NN 50000          // nodes
#define NE 800000         // edges
#define DIM 128
#define OUTC 512          // K|Q|V|S concatenated
#define NG 64             // graphs

typedef __attribute__((ext_vector_type(16))) __bf16 v16bf;
typedef __attribute__((ext_vector_type(8)))  float  v8f;

struct U4 { unsigned x, y, z, w; };                  // POD 16B for b128 loads
union BFrag { v16bf v; unsigned u[8]; U4 q[2]; };

__device__ __forceinline__ unsigned short f32_to_bf16_rne(float f) {
    unsigned u = __float_as_uint(f);
    u += 0x7FFFu + ((u >> 16) & 1u);                 // round-to-nearest-even
    return (unsigned short)(u >> 16);
}

// ---------------------------------------------------------------------------
// Prep: pack Wk|Wq|Wv|Ws (3 layers) transposed into bf16 W4t[L][512][128],
// and biases into b4[L][512].
// ---------------------------------------------------------------------------
__global__ void prep_weights_kernel(const float* __restrict__ Wk, const float* __restrict__ Wq,
                                    const float* __restrict__ Wv, const float* __restrict__ Ws,
                                    const float* __restrict__ bk, const float* __restrict__ bq,
                                    const float* __restrict__ bv, const float* __restrict__ bs,
                                    unsigned short* __restrict__ w4t, float* __restrict__ b4) {
    int i = blockIdx.x * blockDim.x + threadIdx.x;   // 3*512*128
    if (i >= 3 * OUTC * DIM) return;
    int k = i & (DIM - 1);
    int n = (i >> 7) & (OUTC - 1);
    int L = i >> 16;                                 // 512*128 = 65536
    int grp = n >> 7, col = n & (DIM - 1);
    const float* W = (grp == 0) ? Wk : (grp == 1) ? Wq : (grp == 2) ? Wv : Ws;
    w4t[i] = f32_to_bf16_rne(W[L * DIM * DIM + k * DIM + col]);  // transposed: [n][k]
    if (k == 0) {
        const float* B = (grp == 0) ? bk : (grp == 1) ? bq : (grp == 2) ? bv : bs;
        b4[L * OUTC + n] = B[L * DIM + col];
    }
}

// x (fp32) -> xbf (bf16) for layer-0 GEMM input
__global__ void convert_x_kernel(const float* __restrict__ x, unsigned short* __restrict__ xbf) {
    int i = blockIdx.x * blockDim.x + threadIdx.x;
    if (i < NN * DIM) xbf[i] = f32_to_bf16_rne(x[i]);
}

// ---------------------------------------------------------------------------
// WMMA GEMM: out[NN,512] = xbf[NN,128] @ W4t^T + b4.
// Block = 128 threads (4 waves). Block owns a 16-row node tile; wave w owns
// 128 output columns (8 tiles of 16). K=128 -> 4 x v_wmma_f32_16x16x32_bf16.
// A tile (16x128 bf16 = 4KB) is fetched via async global->LDS copy
// (global_load_async_to_lds_b128 + s_wait_asynccnt), bypassing VGPR staging.
// ---------------------------------------------------------------------------
__global__ __launch_bounds__(128)
void gemm_kqvs_kernel(const unsigned short* __restrict__ xbf,
                      const unsigned short* __restrict__ w4t,   // [512][128] bf16 (this layer)
                      const float* __restrict__ b4,             // [512]
                      float* __restrict__ out) {                // [NN][512]
    __shared__ unsigned xlds[16 * DIM / 2];                     // 16x128 bf16 tile (as u32)
    const int tid   = threadIdx.x;
    const int mbase = blockIdx.x * 16;

    // async copy of the A tile: 128 threads x 32B = 4096B
    {
        unsigned lds_addr = (unsigned)(size_t)(&xlds[0]) + tid * 32;       // LDS byte addr
        const unsigned short* gsrc = xbf + (size_t)mbase * DIM + tid * 16; // 32B per thread
        asm volatile("global_load_async_to_lds_b128 %0, %1, off\n\t"
                     "global_load_async_to_lds_b128 %0, %1, off offset:16"
                     :: "v"(lds_addr), "v"(gsrc) : "memory");
        asm volatile("s_wait_asynccnt 0" ::: "memory");
    }
    __syncthreads();

    const int wave  = tid >> 5;
    const int lane  = tid & 31;
    const int half  = lane >> 4;     // 0: lanes 0-15, 1: lanes 16-31
    const int mrow  = lane & 15;
    const int wbase = wave * 128;

    __builtin_prefetch(w4t + wbase * DIM, 0, 0);     // global_prefetch_b8 (weights L2-hot)

    // A fragments (16-bit A 16x32): K = kk*32 + half*16 + {0,2,..,14}(+pair), matching B
    BFrag A[4];
#pragma unroll
    for (int kk = 0; kk < 4; ++kk)
#pragma unroll
        for (int t = 0; t < 8; ++t) {
            int kb = kk * 32 + half * 16 + 2 * t;
            A[kk].u[t] = xlds[(mrow * DIM + kb) >> 1];
        }

#pragma unroll
    for (int nt = 0; nt < 8; ++nt) {
        const int n = wbase + nt * 16 + (lane & 15);
        const float bias = b4[n];
        v8f c = {bias, bias, bias, bias, bias, bias, bias, bias};
#pragma unroll
        for (int kk = 0; kk < 4; ++kk) {
            // B fragment (contiguous k order, same as A): element j -> K = kk*32 + half*16 + j, N = lane&15
            // contiguous in transposed weights => two 16B loads
            BFrag B;
            const U4* p = reinterpret_cast<const U4*>(w4t + n * DIM + kk * 32 + half * 16);
            B.q[0] = p[0];
            B.q[1] = p[1];
            c = __builtin_amdgcn_wmma_f32_16x16x32_bf16(false, A[kk].v, false, B.v,
                                                        (short)0, c, false, false);
        }
        // D layout: VGPR r -> M = r + half*8, N = lane&15
#pragma unroll
        for (int r = 0; r < 8; ++r)
            out[(mbase + r + half * 8) * OUTC + n] = c[r];
    }
}

// ---------------------------------------------------------------------------
// Edge stage: one wave32 per edge; lane covers 4 dims via float4.
// agg[dst] += sigmoid(k[dst] + q[src]) * v[src]   (scatter via f32 atomics; L2-resident)
// ---------------------------------------------------------------------------
__global__ __launch_bounds__(256)
void edge_kernel(const float* __restrict__ kqvs, const int* __restrict__ ei,
                 float* __restrict__ agg) {
    int wid  = (blockIdx.x * blockDim.x + threadIdx.x) >> 5;
    int lane = threadIdx.x & 31;
    if (wid >= NE) return;
    int src = ei[wid];
    int dst = ei[NE + wid];
    int d = lane * 4;
    float4 kd = *(const float4*)(kqvs + dst * OUTC + d);
    float4 qs = *(const float4*)(kqvs + src * OUTC + 128 + d);
    float4 vs = *(const float4*)(kqvs + src * OUTC + 256 + d);
    float4 g;
    g.x = 1.f / (1.f + __expf(-(kd.x + qs.x)));
    g.y = 1.f / (1.f + __expf(-(kd.y + qs.y)));
    g.z = 1.f / (1.f + __expf(-(kd.z + qs.z)));
    g.w = 1.f / (1.f + __expf(-(kd.w + qs.w)));
    float* a = agg + dst * DIM + d;
    atomicAdd(a + 0, g.x * vs.x);
    atomicAdd(a + 1, g.y * vs.y);
    atomicAdd(a + 2, g.z * vs.z);
    atomicAdd(a + 3, g.w * vs.w);
}

// ---------------------------------------------------------------------------
// Node update: y = relu(agg + S), accumulate per-channel sum/sumsq (BN pass 1)
// ---------------------------------------------------------------------------
__global__ __launch_bounds__(128)
void node_update_stats_kernel(const float* __restrict__ agg, const float* __restrict__ kqvs,
                              float* __restrict__ y, float* __restrict__ stats,
                              int rows_per_block) {
    int d  = threadIdx.x;
    int r0 = blockIdx.x * rows_per_block;
    int r1 = r0 + rows_per_block; if (r1 > NN) r1 = NN;
    float s = 0.f, s2 = 0.f;
    for (int n = r0; n < r1; ++n) {
        float v = agg[n * DIM + d] + kqvs[n * OUTC + 384 + d];   // S block
        v = fmaxf(v, 0.f);
        y[n * DIM + d] = v;
        s += v; s2 += v * v;
    }
    atomicAdd(&stats[d], s);
    atomicAdd(&stats[DIM + d], s2);
}

__global__ void bn_finalize_kernel(const float* __restrict__ stats,
                                   const float* __restrict__ g, const float* __restrict__ b,
                                   float* __restrict__ ss, float inv_n) {
    int d = threadIdx.x;                              // 128
    float mu  = stats[d] * inv_n;
    float var = stats[DIM + d] * inv_n - mu * mu;
    float sc  = g[d] * rsqrtf(var + 1e-5f);
    ss[d]       = sc;
    ss[DIM + d] = b[d] - mu * sc;
}

// scale/shift apply; also emits bf16 mirror feeding the next layer's WMMA GEMM
__global__ void bn_apply_kernel(float* __restrict__ y, unsigned short* __restrict__ ybf,
                                const float* __restrict__ ss) {
    int i = blockIdx.x * blockDim.x + threadIdx.x;
    if (i < NN * DIM) {
        int d = i & (DIM - 1);
        float v = y[i] * ss[d] + ss[DIM + d];
        y[i]   = v;
        ybf[i] = f32_to_bf16_rne(v);
    }
}

// ---------------------------------------------------------------------------
// Pooling
// ---------------------------------------------------------------------------
__device__ __forceinline__ void atomicMaxF(float* addr, float val) {
    unsigned* ua = (unsigned*)addr;
    unsigned old = *ua;
    while (__uint_as_float(old) < val) {
        unsigned assumed = old;
        old = atomicCAS(ua, assumed, __float_as_uint(val));
        if (old == assumed) break;
    }
}

__global__ void pool_init_kernel(float* __restrict__ gap, float* __restrict__ gsp,
                                 float* __restrict__ counts) {
    int i = blockIdx.x * blockDim.x + threadIdx.x;
    if (i < NG * DIM) { gap[i] = 0.f; gsp[i] = -3.402823466e38f; }
    if (i < NG) counts[i] = 0.f;
}

__global__ __launch_bounds__(256)
void pool_kernel(const float* __restrict__ x, const int* __restrict__ batch,
                 float* __restrict__ gap, float* __restrict__ gsp, float* __restrict__ counts) {
    int wid  = (blockIdx.x * blockDim.x + threadIdx.x) >> 5;  // node
    int lane = threadIdx.x & 31;
    if (wid >= NN) return;
    int g = batch[wid];
    float4 v = *(const float4*)(x + wid * DIM + lane * 4);
    float* gp = gap + g * DIM + lane * 4;
    atomicAdd(gp + 0, v.x); atomicAdd(gp + 1, v.y);
    atomicAdd(gp + 2, v.z); atomicAdd(gp + 3, v.w);
    float* sp = gsp + g * DIM + lane * 4;
    atomicMaxF(sp + 0, v.x); atomicMaxF(sp + 1, v.y);
    atomicMaxF(sp + 2, v.z); atomicMaxF(sp + 3, v.w);
    if (lane == 0) atomicAdd(&counts[g], 1.f);
}

// mean/where + BN over 64 graphs + concat into feat[64][256]
__global__ void pool_norm_kernel(const float* __restrict__ gap_s, const float* __restrict__ gsp_s,
                                 const float* __restrict__ counts,
                                 const float* __restrict__ g_gap, const float* __restrict__ b_gap,
                                 const float* __restrict__ g_gsp, const float* __restrict__ b_gsp,
                                 float* __restrict__ feat) {
    int c = threadIdx.x;                      // 256
    int is_gsp = c >> 7, d = c & (DIM - 1);
    float vals[NG], s = 0.f, s2 = 0.f;
    for (int g = 0; g < NG; ++g) {
        float cnt = counts[g], v;
        if (!is_gsp) v = gap_s[g * DIM + d] / fmaxf(cnt, 1.f);
        else { v = gsp_s[g * DIM + d]; v = (cnt > 0.f) ? v : 0.f; }
        vals[g] = v; s += v; s2 += v * v;
    }
    float mu = s / (float)NG;
    float istd = rsqrtf(s2 / (float)NG - mu * mu + 1e-5f);
    float gg = is_gsp ? g_gsp[d] : g_gap[d];
    float bb = is_gsp ? b_gsp[d] : b_gap[d];
    for (int g = 0; g < NG; ++g)
        feat[g * 256 + c] = gg * (vals[g] - mu) * istd + bb;
}

// ---------------------------------------------------------------------------
// MLP head (64 rows — tiny)
// ---------------------------------------------------------------------------
__global__ void head_gemm_kernel(const float* __restrict__ in, int in_dim,
                                 const float* __restrict__ W, const float* __restrict__ b,
                                 float* __restrict__ out, int out_dim, int do_relu) {
    int row = blockIdx.x, j = threadIdx.x;
    if (j >= out_dim) return;
    float acc = b[j];
    for (int k = 0; k < in_dim; ++k) acc += in[row * in_dim + k] * W[k * out_dim + j];
    if (do_relu) acc = fmaxf(acc, 0.f);
    out[row * out_dim + j] = acc;
}

__global__ void head_bn_kernel(float* __restrict__ y, int dim,
                               const float* __restrict__ g, const float* __restrict__ b) {
    int c = threadIdx.x;
    if (c >= dim) return;
    float s = 0.f, s2 = 0.f;
    for (int r = 0; r < NG; ++r) { float v = y[r * dim + c]; s += v; s2 += v * v; }
    float mu = s / (float)NG;
    float sc = g[c] * rsqrtf(s2 / (float)NG - mu * mu + 1e-5f);
    float sh = b[c] - mu * sc;
    for (int r = 0; r < NG; ++r) y[r * dim + c] = y[r * dim + c] * sc + sh;
}

// ---------------------------------------------------------------------------
extern "C" void kernel_launch(void* const* d_in, const int* in_sizes, int n_in,
                              void* d_out, int out_size, void* d_ws, size_t ws_size,
                              hipStream_t stream) {
    (void)in_sizes; (void)n_in; (void)out_size; (void)ws_size;
    const float* x0    = (const float*)d_in[0];
    const int*   ei    = (const int*)d_in[1];
    const int*   batch = (const int*)d_in[2];
    const float* Wk = (const float*)d_in[3],  *bk = (const float*)d_in[4];
    const float* Wq = (const float*)d_in[5],  *bq = (const float*)d_in[6];
    const float* Wv = (const float*)d_in[7],  *bv = (const float*)d_in[8];
    const float* Ws = (const float*)d_in[9],  *bs = (const float*)d_in[10];
    const float* g_cl = (const float*)d_in[11], *b_cl = (const float*)d_in[12];
    const float* g_gap = (const float*)d_in[13], *b_gap = (const float*)d_in[14];
    const float* g_gsp = (const float*)d_in[15], *b_gsp = (const float*)d_in[16];
    const float* W1 = (const float*)d_in[17], *b1 = (const float*)d_in[18];
    const float* g1 = (const float*)d_in[19], *bt1 = (const float*)d_in[20];
    const float* W2 = (const float*)d_in[21], *b2 = (const float*)d_in[22];
    const float* g2 = (const float*)d_in[23], *bt2 = (const float*)d_in[24];
    const float* Wl = (const float*)d_in[25], *bl = (const float*)d_in[26];
    float* outp = (float*)d_out;

    // workspace carve-up (256B aligned)
    char* ws = (char*)d_ws;
    size_t off = 0;
    auto alloc = [&](size_t bytes) { void* p = ws + off; off = (off + bytes + 255) & ~(size_t)255; return p; };
    unsigned short* w4t = (unsigned short*)alloc((size_t)3 * OUTC * DIM * 2);  // bf16 weights
    float* b4    = (float*)alloc((size_t)3 * OUTC * 4);
    float* kqvs  = (float*)alloc((size_t)NN * OUTC * 4);   // 102.4 MB
    float* agg   = (float*)alloc((size_t)NN * DIM * 4);    // 25.6 MB
    float* xcur  = (float*)alloc((size_t)NN * DIM * 4);    // 25.6 MB
    unsigned short* xbf = (unsigned short*)alloc((size_t)NN * DIM * 2); // 12.8 MB bf16 mirror
    float* stats = (float*)alloc(2 * DIM * 4);
    float* ss    = (float*)alloc(2 * DIM * 4);
    float* gap   = (float*)alloc((size_t)NG * DIM * 4);
    float* gsp   = (float*)alloc((size_t)NG * DIM * 4);
    float* cnts  = (float*)alloc(NG * 4);
    float* feat  = (float*)alloc((size_t)NG * 256 * 4);
    float* h1    = (float*)alloc((size_t)NG * 256 * 4);
    float* h2    = (float*)alloc((size_t)NG * 128 * 4);

    prep_weights_kernel<<<(3 * OUTC * DIM + 255) / 256, 256, 0, stream>>>(
        Wk, Wq, Wv, Ws, bk, bq, bv, bs, w4t, b4);
    convert_x_kernel<<<(NN * DIM + 255) / 256, 256, 0, stream>>>(x0, xbf);

    for (int L = 0; L < 3; ++L) {
        gemm_kqvs_kernel<<<NN / 16, 128, 0, stream>>>(
            xbf, w4t + (size_t)L * OUTC * DIM, b4 + L * OUTC, kqvs);
        hipMemsetAsync(agg, 0, (size_t)NN * DIM * 4, stream);
        edge_kernel<<<NE / 8, 256, 0, stream>>>(kqvs, ei, agg);
        hipMemsetAsync(stats, 0, 2 * DIM * 4, stream);
        node_update_stats_kernel<<<(NN + 63) / 64, 128, 0, stream>>>(agg, kqvs, xcur, stats, 64);
        bn_finalize_kernel<<<1, 128, 0, stream>>>(stats, g_cl + L * DIM, b_cl + L * DIM,
                                                  ss, 1.0f / (float)NN);
        bn_apply_kernel<<<(NN * DIM + 255) / 256, 256, 0, stream>>>(xcur, xbf, ss);
    }

    pool_init_kernel<<<(NG * DIM + 255) / 256, 256, 0, stream>>>(gap, gsp, cnts);
    pool_kernel<<<NN / 8, 256, 0, stream>>>(xcur, batch, gap, gsp, cnts);
    pool_norm_kernel<<<1, 256, 0, stream>>>(gap, gsp, cnts, g_gap, b_gap, g_gsp, b_gsp, feat);

    head_gemm_kernel<<<NG, 256, 0, stream>>>(feat, 256, W1, b1, h1, 256, 1);
    head_bn_kernel<<<1, 256, 0, stream>>>(h1, 256, g1, bt1);
    head_gemm_kernel<<<NG, 128, 0, stream>>>(h1, 256, W2, b2, h2, 128, 1);
    head_bn_kernel<<<1, 128, 0, stream>>>(h2, 128, g2, bt2);
    head_gemm_kernel<<<NG, 32, 0, stream>>>(h2, 128, Wl, bl, outp, 5, 0);
}
